// NeuronInvariantDeepSetLayer_13872744366612
// MI455X (gfx1250) — matrix operand order repaired and checked by reading, not compiled, not run on hardware
//
#include <hip/hip_runtime.h>
#include <hip/hip_bf16.h>
#include <stdint.h>

// ---------------------------------------------------------------------------
// DeepSets layer for MI455X (gfx1250): split-bf16 WMMA GEMMs + fused sorted
// segment-sum.  x[400000,768] -> phi MLP -> segment_sum[4096,192] -> rho MLP.
// ---------------------------------------------------------------------------

#define N_NODES 400000
#define B_SEG   4096
#define D_IN    768
#define D_HID   192
#define H_PITCH 196   // LDS row pitch (floats) to de-conflict banks

typedef __attribute__((ext_vector_type(16))) __bf16         bf16x16;
typedef __attribute__((ext_vector_type(8)))  float          f32x8;
typedef __attribute__((ext_vector_type(4)))  float          f32x4;
typedef __attribute__((ext_vector_type(8)))  unsigned short u16x8;

union BFrag {
    bf16x16        v;
    unsigned short u[16];
    u16x8          h[2];
};

__device__ __forceinline__ unsigned short f32_to_bf16_rne(float f) {
    unsigned u = __float_as_uint(f);
    u += 0x7FFFu + ((u >> 16) & 1u);        // round-to-nearest-even
    return (unsigned short)(u >> 16);
}
__device__ __forceinline__ float bf16_bits_to_f32(unsigned short s) {
    return __uint_as_float(((unsigned)s) << 16);
}

// ---------------------------------------------------------------------------
// Prep: transpose [K,N] fp32 weights into [N,K] bf16 hi/lo pairs.
// ---------------------------------------------------------------------------
__global__ void conv_weights_kernel(const float* __restrict__ src, int K, int Ncols,
                                    unsigned short* __restrict__ dst_hi,
                                    unsigned short* __restrict__ dst_lo) {
    int e = blockIdx.x * blockDim.x + threadIdx.x;
    if (e >= K * Ncols) return;
    int k = e / Ncols, n = e % Ncols;
    float v = src[e];
    unsigned short hi = f32_to_bf16_rne(v);
    float residual = v - bf16_bits_to_f32(hi);
    dst_hi[(size_t)n * K + k] = hi;
    dst_lo[(size_t)n * K + k] = f32_to_bf16_rne(residual);
}

// ---------------------------------------------------------------------------
// Fused phi MLP + segment scatter-add.
// 128 threads = 4 waves; wave w owns rows [ (blockIdx*4+w)*16 , +16 ).
// ---------------------------------------------------------------------------
__global__ __launch_bounds__(128, 1) void deepset_phi_kernel(
    const float* __restrict__ x, const int* __restrict__ idx,
    const unsigned short* __restrict__ w1t_hi, const unsigned short* __restrict__ w1t_lo,
    const float* __restrict__ b1,
    const unsigned short* __restrict__ w2t_hi, const unsigned short* __restrict__ w2t_lo,
    const float* __restrict__ b2,
    float* __restrict__ xsum) {

    __shared__ float lds_h1[4][16 * H_PITCH];   // 50,176 B, wave-private slabs

    const int  lane    = threadIdx.x & 31;
    const int  wave    = threadIdx.x >> 5;
    const int  rowTile = (blockIdx.x * 4 + wave) * 16;
    const int  lm      = lane & 15;
    const bool hiHalf  = lane >= 16;
    const int  kA_off  = hiHalf ? 8  : 0;   // A frag: K blocks {0..7,16..23} / {8..15,24..31}
    const int  kB_off  = hiHalf ? 8  : 0;   // B frag: K blocks {0..7,16..23} / {8..15,24..31}
    const int  rowOff  = hiHalf ? 8  : 0;   // C frag: rows r / r+8

    f32x8 acc[12];
    #pragma unroll
    for (int t = 0; t < 12; ++t)
        #pragma unroll
        for (int i = 0; i < 8; ++i) acc[t][i] = 0.0f;

    // ------------------ GEMM1: h1 = x @ W1  (K = 768) ------------------
    const float* xrow = x + (size_t)(rowTile + lm) * D_IN;
    #pragma unroll 1
    for (int kc = 0; kc < D_IN; kc += 32) {
        BFrag Ahi, Alo;
        f32x4 a0 = *(const f32x4*)(xrow + kc + kA_off);
        f32x4 a1 = *(const f32x4*)(xrow + kc + kA_off + 4);
        f32x4 a2 = *(const f32x4*)(xrow + kc + kA_off + 16);
        f32x4 a3 = *(const f32x4*)(xrow + kc + kA_off + 20);
        float av[16];
        #pragma unroll
        for (int i = 0; i < 4; ++i) { av[i]=a0[i]; av[4+i]=a1[i]; av[8+i]=a2[i]; av[12+i]=a3[i]; }
        #pragma unroll
        for (int i = 0; i < 16; ++i) {
            unsigned short h = f32_to_bf16_rne(av[i]);
            Ahi.u[i] = h;
            Alo.u[i] = f32_to_bf16_rne(av[i] - bf16_bits_to_f32(h));
        }
        #pragma unroll
        for (int t = 0; t < 12; ++t) {
            const size_t boff = (size_t)(t * 16 + lm) * D_IN + kc + kB_off;
            BFrag Bhi, Blo;
            Bhi.h[0] = *(const u16x8*)(w1t_hi + boff);
            Bhi.h[1] = *(const u16x8*)(w1t_hi + boff + 16);
            Blo.h[0] = *(const u16x8*)(w1t_lo + boff);
            Blo.h[1] = *(const u16x8*)(w1t_lo + boff + 16);
            acc[t] = __builtin_amdgcn_wmma_f32_16x16x32_bf16(false, Ahi.v, false, Bhi.v, (short)0, acc[t], false, false);
            acc[t] = __builtin_amdgcn_wmma_f32_16x16x32_bf16(false, Ahi.v, false, Blo.v, (short)0, acc[t], false, false);
            acc[t] = __builtin_amdgcn_wmma_f32_16x16x32_bf16(false, Alo.v, false, Bhi.v, (short)0, acc[t], false, false);
        }
    }

    // bias + ReLU, stage h1 (fp32) into wave-private LDS in [row][k] order
    float* myH = &lds_h1[wave][0];
    #pragma unroll
    for (int t = 0; t < 12; ++t) {
        int col = t * 16 + lm;
        float bias = b1[col];
        #pragma unroll
        for (int r = 0; r < 8; ++r) {
            float v = acc[t][r] + bias;
            myH[(rowOff + r) * H_PITCH + col] = v > 0.0f ? v : 0.0f;
        }
    }
    asm volatile("s_wait_dscnt 0x0" ::: "memory");   // intra-wave cross-lane LDS RAW

    // ------------------ GEMM2: h2 = h1 @ W2  (K = 192) ------------------
    #pragma unroll
    for (int t = 0; t < 12; ++t)
        #pragma unroll
        for (int i = 0; i < 8; ++i) acc[t][i] = 0.0f;

    #pragma unroll 1
    for (int kc = 0; kc < D_HID; kc += 32) {
        BFrag Ahi, Alo;
        const float* hp = myH + lm * H_PITCH + kc + kA_off;
        f32x4 a0 = *(const f32x4*)(hp);
        f32x4 a1 = *(const f32x4*)(hp + 4);
        f32x4 a2 = *(const f32x4*)(hp + 16);
        f32x4 a3 = *(const f32x4*)(hp + 20);
        float av[16];
        #pragma unroll
        for (int i = 0; i < 4; ++i) { av[i]=a0[i]; av[4+i]=a1[i]; av[8+i]=a2[i]; av[12+i]=a3[i]; }
        #pragma unroll
        for (int i = 0; i < 16; ++i) {
            unsigned short h = f32_to_bf16_rne(av[i]);
            Ahi.u[i] = h;
            Alo.u[i] = f32_to_bf16_rne(av[i] - bf16_bits_to_f32(h));
        }
        #pragma unroll
        for (int t = 0; t < 12; ++t) {
            const size_t boff = (size_t)(t * 16 + lm) * D_HID + kc + kB_off;
            BFrag Bhi, Blo;
            Bhi.h[0] = *(const u16x8*)(w2t_hi + boff);
            Bhi.h[1] = *(const u16x8*)(w2t_hi + boff + 16);
            Blo.h[0] = *(const u16x8*)(w2t_lo + boff);
            Blo.h[1] = *(const u16x8*)(w2t_lo + boff + 16);
            acc[t] = __builtin_amdgcn_wmma_f32_16x16x32_bf16(false, Ahi.v, false, Bhi.v, (short)0, acc[t], false, false);
            acc[t] = __builtin_amdgcn_wmma_f32_16x16x32_bf16(false, Ahi.v, false, Blo.v, (short)0, acc[t], false, false);
            acc[t] = __builtin_amdgcn_wmma_f32_16x16x32_bf16(false, Alo.v, false, Bhi.v, (short)0, acc[t], false, false);
        }
    }

    // bias + sorted-segment scatter: run-length compress the 8 rows per lane
    int segv[8];
    #pragma unroll
    for (int r = 0; r < 8; ++r) segv[r] = idx[rowTile + rowOff + r];

    #pragma unroll
    for (int t = 0; t < 12; ++t) {
        int col = t * 16 + lm;
        float bias = b2[col];
        float run = 0.0f;
        int cur = segv[0];
        #pragma unroll
        for (int r = 0; r < 8; ++r) {
            if (segv[r] != cur) {
                atomicAdd(xsum + (size_t)cur * D_HID + col, run);
                run = 0.0f;
                cur = segv[r];
            }
            run += acc[t][r] + bias;
        }
        atomicAdd(xsum + (size_t)cur * D_HID + col, run);
    }
}

// ---------------------------------------------------------------------------
// rho MLP: [4096,192] -> relu(@[192,6]+b) -> @[6,1]+b -> [4096]
// ---------------------------------------------------------------------------
__global__ void rho_kernel(const float* __restrict__ xsum,
                           const float* __restrict__ rw1, const float* __restrict__ rb1,
                           const float* __restrict__ rw2, const float* __restrict__ rb2,
                           float* __restrict__ out) {
    int b = blockIdx.x * blockDim.x + threadIdx.x;
    if (b >= B_SEG) return;
    const float* xr = xsum + (size_t)b * D_HID;
    float r[6];
    #pragma unroll
    for (int j = 0; j < 6; ++j) r[j] = rb1[j];
    for (int k = 0; k < D_HID; ++k) {
        float v = xr[k];
        #pragma unroll
        for (int j = 0; j < 6; ++j) r[j] = fmaf(v, rw1[k * 6 + j], r[j]);
    }
    float o = rb2[0];
    #pragma unroll
    for (int j = 0; j < 6; ++j) {
        float a = r[j] > 0.0f ? r[j] : 0.0f;
        o = fmaf(a, rw2[j], o);
    }
    out[b] = o;
}

// ---------------------------------------------------------------------------
extern "C" void kernel_launch(void* const* d_in, const int* in_sizes, int n_in,
                              void* d_out, int out_size, void* d_ws, size_t ws_size,
                              hipStream_t stream) {
    const float* x      = (const float*)d_in[0];
    const int*   idx    = (const int*)  d_in[1];
    const float* phi_w1 = (const float*)d_in[2];
    const float* phi_b1 = (const float*)d_in[3];
    const float* phi_w2 = (const float*)d_in[4];
    const float* phi_b2 = (const float*)d_in[5];
    const float* rho_w1 = (const float*)d_in[6];
    const float* rho_b1 = (const float*)d_in[7];
    const float* rho_w2 = (const float*)d_in[8];
    const float* rho_b2 = (const float*)d_in[9];
    float* out = (float*)d_out;

    // workspace layout
    char* ws = (char*)d_ws;
    unsigned short* w1t_hi = (unsigned short*)ws;                 // 192*768
    unsigned short* w1t_lo = w1t_hi + (size_t)D_HID * D_IN;       // 192*768
    unsigned short* w2t_hi = w1t_lo + (size_t)D_HID * D_IN;       // 192*192
    unsigned short* w2t_lo = w2t_hi + (size_t)D_HID * D_HID;      // 192*192
    float* xsum = (float*)(ws + 737280);                          // 4096*192 f32

    hipMemsetAsync(xsum, 0, (size_t)B_SEG * D_HID * sizeof(float), stream);

    conv_weights_kernel<<<(D_IN * D_HID + 255) / 256, 256, 0, stream>>>(
        phi_w1, D_IN, D_HID, w1t_hi, w1t_lo);
    conv_weights_kernel<<<(D_HID * D_HID + 255) / 256, 256, 0, stream>>>(
        phi_w2, D_HID, D_HID, w2t_hi, w2t_lo);

    // 400000 rows = 6250 blocks * 4 waves * 16 rows (exact, no tail)
    deepset_phi_kernel<<<N_NODES / 64, 128, 0, stream>>>(
        x, idx, w1t_hi, w1t_lo, phi_b1, w2t_hi, w2t_lo, phi_b2, xsum);

    rho_kernel<<<(B_SEG + 127) / 128, 128, 0, stream>>>(
        xsum, rho_w1, rho_b1, rho_w2, rho_b2, out);
}
